// WaveletNet_2293512536286
// MI455X (gfx1250) — matrix operand order repaired and checked by reading, not compiled, not run on hardware
//
#include <hip/hip_runtime.h>
#include <hip/hip_bf16.h>

typedef __attribute__((ext_vector_type(16))) __bf16 v16bf;
typedef __attribute__((ext_vector_type(8)))  float  v8f;

#define NPIX    1048576.0f   // 16*256*256 elements per channel for BN stats
#define HI_ELEM 67108864ull  // 16*64*256*256

// ---------------------------------------------------------------------------
// Kernel 1: repack weights into WMMA B-fragment layout (bf16), zero the stats.
// B fragment layout chosen here for v_wmma_f32_16x16x32_bf16 (32x16, K x N), contiguous K order, same as A:
//   lane L: N = L&15 ; K-half = (L>>4)*16 ; element j (0..15): k = K-half + j
// Stored so each lane's 16 bf16 are contiguous: wb[((ks*4+nt)*32+lane)*16+j]
// K layout: k = ks*32 + (c*9 + r*3 + q), padded 27->32 per 32-wide K step.
// high: ic = ks*3 + c (concat [LH,HL,HH]) ; low: single K step, ic = c.
// ---------------------------------------------------------------------------
__global__ void prep_weights(const float* __restrict__ hw,
                             const float* __restrict__ lw,
                             __bf16* __restrict__ wbHi,
                             __bf16* __restrict__ wbLo,
                             float*  __restrict__ sums) {
    int e = blockIdx.x * 256 + threadIdx.x;   // 8192 total entries
    if (e < 256) sums[e] = 0.0f;              // zero stat accumulators
    if (e < 6144) {                           // high: 3 K-steps * 4 n-tiles * 32 lanes * 16
        int j    = e & 15;
        int lane = (e >> 4) & 31;
        int nt   = (e >> 9) & 3;
        int ks   = e >> 11;
        int n    = nt * 16 + (lane & 15);
        int ksub = (lane >> 4) * 16 + j;      // 0..31 within K-step
        float v = 0.0f;
        if (ksub < 27) {
            int c = ksub / 9, rem = ksub % 9, r = rem / 3, q = rem % 3;
            int ic = ks * 3 + c;
            v = hw[((n * 9 + ic) * 3 + r) * 3 + q];
        }
        wbHi[e] = (__bf16)v;
    } else if (e < 8192) {                    // low: 1 K-step
        int e2   = e - 6144;
        int j    = e2 & 15;
        int lane = (e2 >> 4) & 31;
        int nt   = (e2 >> 9) & 3;
        int n    = nt * 16 + (lane & 15);
        int ksub = (lane >> 4) * 16 + j;
        float v = 0.0f;
        if (ksub < 27) {
            int c = ksub / 9, rem = ksub % 9, r = rem / 3, q = rem % 3;
            v = lw[((n * 3 + c) * 3 + r) * 3 + q];
        }
        wbLo[e2] = (__bf16)v;
    }
}

// ---------------------------------------------------------------------------
// Kernel 2: fused Haar-DWT + implicit-GEMM conv3x3 on WMMA bf16.
// Block = 256 threads (8 waves), tile = (batch b, half-res row y, 128 cols).
// Two instantiations:
//   APPLY=false : statistics pass — reduce sum/sum^2 straight from the WMMA
//                 accumulators (LDS ds_add_f32 + 1 global atomic/ch/block),
//                 zero global stores.
//   APPLY=true  : recompute conv, fuse per-channel scale/bias + ReLU on the
//                 accumulators, LDS transpose, coalesced float4 stores.
// Recompute beats re-reading 536 MB: total traffic ~0.84 GB vs ~1.75 GB.
// ---------------------------------------------------------------------------
template <bool APPLY>
__global__ __launch_bounds__(256) void conv_wmma(const float* __restrict__ in,
                                                 const __bf16* __restrict__ wbHi,
                                                 const __bf16* __restrict__ wbLo,
                                                 float* __restrict__ out,
                                                 float* __restrict__ sums,
                                                 const float* __restrict__ sb) {
    __shared__ __bf16 subb[4][3][3][132];   // [LL,LH,HL,HH][c][row-1..+1][col x0-1..x0+128]
    __shared__ float  outtile[64][132];     // APPLY only: bank-conflict-free transpose
    __shared__ float  bsum[128];            // stats only: [0,64) sum, [64,128) sum^2

    const int x0 = blockIdx.x * 128;
    const int y  = blockIdx.y;
    const int b  = blockIdx.z;

    // ---- Phase 1: Haar DWT of the halo tile straight into LDS (bf16) ------
    for (int p = threadIdx.x; p < 3 * 3 * 130; p += 256) {
        int c   = p / 390;
        int rr  = (p / 130) % 3;
        int col = p % 130;
        int Y = y - 1 + rr;
        int X = x0 - 1 + col;
        float ll = 0.f, lh = 0.f, hl = 0.f, hh = 0.f;
        if (Y >= 0 && Y < 256 && X >= 0 && X < 256) {
            const float2* r0 = (const float2*)(in + (((size_t)b * 3 + c) * 512 + 2 * Y) * 512);
            const float2* r1 = (const float2*)(in + (((size_t)b * 3 + c) * 512 + 2 * Y + 1) * 512);
            float2 p0 = r0[X];
            float2 p1 = r1[X];
            ll = (p0.x + p0.y + p1.x + p1.y) * 0.5f;
            lh = (p0.x + p0.y - p1.x - p1.y) * 0.5f;
            hl = (p0.x - p0.y + p1.x - p1.y) * 0.5f;
            hh = (p0.x - p0.y - p1.x + p1.y) * 0.5f;
        }
        subb[0][c][rr][col] = (__bf16)ll;
        subb[1][c][rr][col] = (__bf16)lh;
        subb[2][c][rr][col] = (__bf16)hl;
        subb[3][c][rr][col] = (__bf16)hh;
    }
    __syncthreads();

    const int lane    = threadIdx.x & 31;
    const int wave    = threadIdx.x >> 5;
    const int m       = lane & 15;      // A-matrix row  (pixel within wave tile)
    const int hi      = lane >> 4;      // K-half selector per ISA A layout
    const int px_base = wave * 16;
    const int px      = px_base + m;

    // A fragment (16x32 bf16) gathered from one LDS subband, K padded 27->32.
    // element j: k = hi*16+j  [contiguous K order, matches B fragment]
    auto loadA = [&](int s) -> v16bf {
        v16bf a;
        #pragma unroll
        for (int j = 0; j < 16; ++j) {
            int kl = hi * 16 + j;
            __bf16 v = (__bf16)0.0f;
            if (kl < 27) {
                int c = kl / 9, rem = kl % 9, r = rem / 3, q = rem % 3;
                v = subb[s][c][r][px + q];
            }
            a[j] = v;
        }
        return a;
    };

    // Consume one branch's accumulators (D layout: VGPR r -> pixel hi*8+r,
    // lane&15 -> channel-in-tile).
    auto consume = [&](v8f (&acc)[4], float* outBase, int branchBase) {
        if constexpr (APPLY) {
            #pragma unroll
            for (int nt = 0; nt < 4; ++nt) {
                int ch = nt * 16 + m;
                float sc = sb[branchBase + ch];
                float bi = sb[branchBase + 64 + ch];
                #pragma unroll
                for (int r = 0; r < 8; ++r)
                    outtile[ch][px_base + hi * 8 + r] =
                        fmaxf(fmaf(acc[nt][r], sc, bi), 0.0f);
            }
            __syncthreads();
            int t = threadIdx.x;
            #pragma unroll
            for (int i = 0; i < 8; ++i) {
                int lin = t + i * 256;            // 2048 float4 slots = 64ch * 32
                int ch  = lin >> 5;
                int p4  = lin & 31;
                float4 v = make_float4(outtile[ch][p4 * 4 + 0], outtile[ch][p4 * 4 + 1],
                                       outtile[ch][p4 * 4 + 2], outtile[ch][p4 * 4 + 3]);
                *(float4*)(outBase + (size_t)ch * 65536 + x0 + p4 * 4) = v;
            }
            __syncthreads();
        } else {
            if (threadIdx.x < 128) bsum[threadIdx.x] = 0.0f;
            __syncthreads();
            #pragma unroll
            for (int nt = 0; nt < 4; ++nt) {
                int ch = nt * 16 + m;
                float s = 0.0f, s2 = 0.0f;
                #pragma unroll
                for (int r = 0; r < 8; ++r) {
                    float v = acc[nt][r];
                    s += v; s2 += v * v;
                }
                atomicAdd(&bsum[ch], s);          // ds_add_f32
                atomicAdd(&bsum[64 + ch], s2);
            }
            __syncthreads();
            if (threadIdx.x < 128)
                atomicAdd(&sums[branchBase + threadIdx.x], bsum[threadIdx.x]);
            __syncthreads();
        }
    };

    const v8f vzero = {0.f, 0.f, 0.f, 0.f, 0.f, 0.f, 0.f, 0.f};
    float* hiBase = out + (size_t)b * 64 * 65536 + (size_t)y * 256;
    float* loBase = out + HI_ELEM + (size_t)b * 64 * 65536 + (size_t)y * 256;

    // ---- Phase 2: high branch — K = 96 (3 steps), N = 64 (4 tiles) --------
    {
        v8f acc[4] = {vzero, vzero, vzero, vzero};
        #pragma unroll
        for (int ks = 0; ks < 3; ++ks) {
            v16bf a = loadA(1 + ks);    // subbands LH, HL, HH
            #pragma unroll
            for (int nt = 0; nt < 4; ++nt) {
                v16bf bf = *(const v16bf*)(wbHi + ((ks * 4 + nt) * 32 + lane) * 16);
                acc[nt] = __builtin_amdgcn_wmma_f32_16x16x32_bf16(
                    false, a, false, bf, (short)0, acc[nt], false, false);
            }
        }
        consume(acc, hiBase, 0);
    }

    // ---- Phase 3: low branch — K = 32 (1 step) ----------------------------
    {
        v8f acc[4] = {vzero, vzero, vzero, vzero};
        v16bf a = loadA(0);             // subband LL
        #pragma unroll
        for (int nt = 0; nt < 4; ++nt) {
            v16bf bf = *(const v16bf*)(wbLo + (nt * 32 + lane) * 16);
            acc[nt] = __builtin_amdgcn_wmma_f32_16x16x32_bf16(
                false, a, false, bf, (short)0, acc[nt], false, false);
        }
        consume(acc, loBase, 128);
    }
}

// ---------------------------------------------------------------------------
// Kernel 3: sums -> per-channel scale/bias.  (Conv bias cancels in BN: a
// per-channel constant shifts the mean identically, so it is omitted.)
// ---------------------------------------------------------------------------
__global__ void finalize_stats(const float* __restrict__ sums,
                               const float* __restrict__ hg, const float* __restrict__ hbe,
                               const float* __restrict__ lg, const float* __restrict__ lbe,
                               float* __restrict__ sb) {
    int t = threadIdx.x;              // 128 threads: [0,64) high, [64,128) low
    if (t >= 128) return;
    int branch = t >> 6, ch = t & 63;
    float s  = sums[branch * 128 + ch];
    float s2 = sums[branch * 128 + 64 + ch];
    float mean = s / NPIX;
    float var  = s2 / NPIX - mean * mean;
    float g  = branch ? lg[ch]  : hg[ch];
    float be = branch ? lbe[ch] : hbe[ch];
    float sc = g * rsqrtf(var + 1e-5f);
    sb[branch * 128 + ch]      = sc;
    sb[branch * 128 + 64 + ch] = be - mean * sc;
}

// ---------------------------------------------------------------------------
extern "C" void kernel_launch(void* const* d_in, const int* in_sizes, int n_in,
                              void* d_out, int out_size, void* d_ws, size_t ws_size,
                              hipStream_t stream) {
    const float* x     = (const float*)d_in[0];
    const float* hw    = (const float*)d_in[1];
    // d_in[2] = high_b : cancelled by batchnorm mean subtraction
    const float* hg    = (const float*)d_in[3];
    const float* hbeta = (const float*)d_in[4];
    const float* lw    = (const float*)d_in[5];
    // d_in[6] = low_b  : cancelled by batchnorm mean subtraction
    const float* lg    = (const float*)d_in[7];
    const float* lbeta = (const float*)d_in[8];

    char* ws = (char*)d_ws;
    __bf16* wbHi = (__bf16*)(ws);            // 6144 bf16 = 12288 B
    __bf16* wbLo = (__bf16*)(ws + 12288);    // 2048 bf16 =  4096 B
    float*  sums = (float*)(ws + 16384);     // 256 fp32 (hiSum,hiSq,loSum,loSq)
    float*  sb   = (float*)(ws + 17408);     // 256 fp32 (scale/bias both branches)

    prep_weights<<<32, 256, 0, stream>>>(hw, lw, wbHi, wbLo, sums);

    dim3 grid(2, 256, 16);                   // x-tile, half-res row, batch

    // Pass 1: statistics only (no global output traffic).
    conv_wmma<false><<<grid, 256, 0, stream>>>(x, wbHi, wbLo, (float*)d_out, sums, sb);

    finalize_stats<<<1, 128, 0, stream>>>(sums, hg, hbeta, lg, lbeta, sb);

    // Pass 2: recompute conv, fused BN + ReLU, single 536 MB store.
    conv_wmma<true><<<grid, 256, 0, stream>>>(x, wbHi, wbLo, (float*)d_out, sums, sb);
}
